// AFP_GATGRUConvMol_47605417509079
// MI455X (gfx1250) — matrix operand order repaired and checked by reading, not compiled, not run on hardware
//
#include <hip/hip_runtime.h>
#include <hip/hip_bf16.h>

// ---------- constants from the reference ----------
#define N_SRC 100000
#define N_DST 8192
#define NEDGE 100000
#define D_IN  256
#define HC    1024      // H*C
#define G3    768       // 3*D_GRU
#define NEG_SLOPE 0.2f

typedef __attribute__((ext_vector_type(16))) __bf16 v16bf;
typedef __attribute__((ext_vector_type(8)))  float  v8f;

union Frag16 { v16bf v; unsigned u[8]; };
union Frag8F { v8f v; float f[8]; };

static __device__ __forceinline__ unsigned short f2bf(float f) {
    unsigned u = __float_as_uint(f);
    unsigned r = u + 0x7FFFu + ((u >> 16) & 1u);   // round-to-nearest-even
    return (unsigned short)(r >> 16);
}
static __device__ __forceinline__ unsigned pack2bf(float lo, float hi) {
    return (unsigned)f2bf(lo) | ((unsigned)f2bf(hi) << 16);
}
static __device__ __forceinline__ float bf2f(unsigned short h) {
    return __uint_as_float(((unsigned)h) << 16);
}
static __device__ __forceinline__ void atomicMaxF(float* addr, float val) {
    if (val >= 0.0f) atomicMax((int*)addr, __float_as_int(val));
    else             atomicMin((unsigned*)addr, __float_as_uint(val));
}
static __device__ __forceinline__ float sigmoidf(float x) {
    return 1.0f / (1.0f + __expf(-x));
}

// =====================================================================
// Tiled WMMA GEMM:  C[M,N] = A[M,K] * B + bias[N]
//   A_BF16:   A stored as bf16 (else f32, converted on LDS stage)
//   B_TRANS:  B given row-major [N,K] (C = A*B^T), else row-major [K,N]
//   OUT_BF16: store C as bf16 (else f32)
// Block: 256 threads (8 waves). Block tile 64(M) x 256(N), K-step 32.
// Waves arranged 2(M) x 4(N); each wave owns 32x64 = 8 accumulators:
// 2 A-fragments reused across 4 B-fragments -> 8 WMMAs per K-step.
// =====================================================================
#define TM 64
#define TN 256
#define TK 32

template<bool A_BF16, bool B_TRANS, bool OUT_BF16>
__global__ __launch_bounds__(256)
void gemm_tile(const void* __restrict__ Av, const float* __restrict__ Bw,
               const float* __restrict__ bias, void* __restrict__ Cv,
               int M, int N, int K)
{
    __shared__ unsigned short As[TM][TK + 2];   // pad 2 -> 68B row stride
    __shared__ unsigned short Bs[TN][TK + 2];   // n-major: Bs[n][k]

    const int tid  = threadIdx.x;
    const int lane = tid & 31;
    const int w    = tid >> 5;
    const int wmv  = (w & 1) * 32;    // wave M offset in tile
    const int wnv  = (w >> 1) * 64;   // wave N offset in tile
    const int hf   = lane >> 4;       // lane half
    const int l16  = lane & 15;

    const int m0 = blockIdx.x * TM;
    const int n0 = blockIdx.y * TN;

    const float*          Af = (const float*)Av;
    const unsigned short* Ab = (const unsigned short*)Av;

    v8f acc[2][4];
    #pragma unroll
    for (int s = 0; s < 2; ++s)
        #pragma unroll
        for (int t = 0; t < 4; ++t) acc[s][t] = (v8f){};

    for (int kb = 0; kb < K; kb += TK) {
        // ---- stage A tile (64x32): 8 elems / thread, 32-bit LDS stores
        {
            int row = tid >> 2;
            int col = (tid & 3) * 8;
            int gm  = m0 + row;
            unsigned* dstp = (unsigned*)&As[row][col];
            if (gm < M) {
                if (A_BF16) {
                    const unsigned* p = (const unsigned*)(Ab + (size_t)gm * K + kb + col);
                    #pragma unroll
                    for (int j = 0; j < 4; ++j) dstp[j] = p[j];
                } else {
                    const float* p = Af + (size_t)gm * K + kb + col;
                    #pragma unroll
                    for (int j = 0; j < 4; ++j) dstp[j] = pack2bf(p[2 * j], p[2 * j + 1]);
                }
            } else {
                #pragma unroll
                for (int j = 0; j < 4; ++j) dstp[j] = 0u;
            }
            // prefetch next A tile into near cache (global_prefetch_b8)
            if (kb + TK < K && gm < M) {
                if (A_BF16) __builtin_prefetch(Ab + (size_t)gm * K + kb + TK + col, 0, 3);
                else        __builtin_prefetch(Af + (size_t)gm * K + kb + TK + col, 0, 3);
            }
        }
        // ---- stage B tile (256 n x 32 k) into Bs[n][k]: 32 elems / thread
        if (B_TRANS) {
            int n = tid;                               // one row of W[N][K]
            const float* p = Bw + (size_t)(n0 + n) * K + kb;
            unsigned* dstp = (unsigned*)&Bs[n][0];
            #pragma unroll
            for (int j = 0; j < 16; ++j) dstp[j] = pack2bf(p[2 * j], p[2 * j + 1]);
        } else {
            int k  = tid >> 3;                         // 0..31
            int nc = (tid & 7) * 32;                   // 32 consecutive n
            const float* p = Bw + (size_t)(kb + k) * N + n0 + nc;
            #pragma unroll
            for (int j = 0; j < 32; ++j) Bs[nc + j][k] = f2bf(p[j]);
        }
        __syncthreads();

        // ---- build fragments (ISA 7.12.2 16-bit layouts) and 8 WMMAs
        Frag16 a[2];
        #pragma unroll
        for (int s = 0; s < 2; ++s) {
            int row = wmv + 16 * s + l16;
            #pragma unroll
            for (int i = 0; i < 8; ++i) {
                int j  = 2 * i;
                int kk = (j < 8 ? j : j + 8) + 8 * hf; // pairs: 0-7/16-23 | 8-15/24-31
                a[s].u[i] = *(const unsigned*)&As[row][kk];
            }
        }
        #pragma unroll
        for (int t = 0; t < 4; ++t) {
            Frag16 b;
            int nc0 = wnv + 16 * t + l16;
            #pragma unroll
            for (int i = 0; i < 8; ++i) {
                int kk = (2 * i < 8 ? 2 * i : 2 * i + 8) + 8 * hf; // pairs: 0-7/16-23 | 8-15/24-31
                b.u[i] = *(const unsigned*)&Bs[nc0][kk];
            }
            acc[0][t] = __builtin_amdgcn_wmma_f32_16x16x32_bf16(false, a[0].v, false, b.v,
                                                                (short)0, acc[0][t], false, false);
            acc[1][t] = __builtin_amdgcn_wmma_f32_16x16x32_bf16(false, a[1].v, false, b.v,
                                                                (short)0, acc[1][t], false, false);
        }
        __syncthreads();
    }

    // ---- epilogue: + bias, store
    float*          Cf = (float*)Cv;
    unsigned short* Cb = (unsigned short*)Cv;
    #pragma unroll
    for (int s = 0; s < 2; ++s) {
        #pragma unroll
        for (int t = 0; t < 4; ++t) {
            Frag8F r; r.v = acc[s][t];
            int gn = n0 + wnv + 16 * t + l16;
            float bv = bias[gn];
            #pragma unroll
            for (int rr = 0; rr < 8; ++rr) {
                int gm = m0 + wmv + 16 * s + rr + 8 * hf;
                if (gm < M) {
                    float v = r.f[rr] + bv;
                    if (OUT_BF16) Cb[(size_t)gm * N + gn] = f2bf(v);
                    else          Cf[(size_t)gm * N + gn] = v;
                }
            }
        }
    }
}

// =====================================================================
// workspace init: agg = 0, m = -inf, denom = 0
// =====================================================================
__global__ void init_ws_kernel(float* mbuf, float* denom, float* agg,
                               int nAgg, int nMD)
{
    int t = blockIdx.x * blockDim.x + threadIdx.x;
    if (t < nAgg) agg[t] = 0.0f;
    if (t < nMD) { mbuf[t] = __uint_as_float(0xFF800000u); denom[t] = 0.0f; }
}

// =====================================================================
// per-edge attention logits + segment max (one wave per edge)
// =====================================================================
__global__ __launch_bounds__(256)
void edge_logits_kernel(const unsigned short* __restrict__ xl,
                        const unsigned short* __restrict__ xr,
                        const int* __restrict__ src, const int* __restrict__ dst,
                        const float* __restrict__ att,
                        float* __restrict__ logits, float* __restrict__ mbuf, int E)
{
    int e = blockIdx.x * 8 + (threadIdx.x >> 5);
    if (e >= E) return;
    int lane = threadIdx.x & 31;
    int s = src[e], d = dst[e];
    const unsigned* pl = (const unsigned*)(xl + (size_t)s * HC);
    const unsigned* pr = (const unsigned*)(xr + (size_t)d * HC);
    float acc[4] = {0.f, 0.f, 0.f, 0.f};
    #pragma unroll
    for (int it = 0; it < 16; ++it) {
        int idx2 = it * 32 + lane;              // bf16-pair index, head = it/4
        unsigned a = pl[idx2], b = pr[idx2];
        float x0 = bf2f((unsigned short)(a & 0xFFFFu)) + bf2f((unsigned short)(b & 0xFFFFu));
        float x1 = bf2f((unsigned short)(a >> 16))     + bf2f((unsigned short)(b >> 16));
        x0 = x0 > 0.0f ? x0 : NEG_SLOPE * x0;
        x1 = x1 > 0.0f ? x1 : NEG_SLOPE * x1;
        acc[it >> 2] += x0 * att[2 * idx2] + x1 * att[2 * idx2 + 1];
    }
    #pragma unroll
    for (int h = 0; h < 4; ++h) {
        float v = acc[h];
        for (int off = 16; off > 0; off >>= 1) v += __shfl_xor(v, off, 32);
        if (lane == 0) {
            logits[e * 4 + h] = v;
            atomicMaxF(&mbuf[d * 4 + h], v);
        }
    }
}

// =====================================================================
// a = exp(logit - m[dst]); denom[dst] += a   (one thread per edge*head)
// =====================================================================
__global__ void edge_exp_kernel(const int* __restrict__ dst,
                                const float* __restrict__ mbuf,
                                float* __restrict__ logits,
                                float* __restrict__ denom, int E4)
{
    int t = blockIdx.x * blockDim.x + threadIdx.x;
    if (t >= E4) return;
    int e = t >> 2, h = t & 3;
    int d = dst[e];
    float a = __expf(logits[t] - mbuf[d * 4 + h]);
    logits[t] = a;
    atomicAdd(&denom[d * 4 + h], a);
}

// =====================================================================
// agg[dst] += xl[src] * alpha  (one wave per edge, f32 atomics in L2)
// =====================================================================
__global__ __launch_bounds__(256)
void edge_agg_kernel(const unsigned short* __restrict__ xl,
                     const int* __restrict__ src, const int* __restrict__ dst,
                     const float* __restrict__ aexp, const float* __restrict__ denom,
                     float* __restrict__ agg, int E)
{
    int e = blockIdx.x * 8 + (threadIdx.x >> 5);
    if (e >= E) return;
    int lane = threadIdx.x & 31;
    int s = src[e], d = dst[e];
    float alpha[4];
    #pragma unroll
    for (int h = 0; h < 4; ++h)
        alpha[h] = aexp[e * 4 + h] / (denom[d * 4 + h] + 1e-16f);
    const unsigned* pl = (const unsigned*)(xl + (size_t)s * HC);
    float* po = agg + (size_t)d * HC;
    #pragma unroll
    for (int it = 0; it < 16; ++it) {
        int idx2 = it * 32 + lane;
        unsigned a = pl[idx2];
        float al = alpha[it >> 2];
        atomicAdd(&po[2 * idx2],     bf2f((unsigned short)(a & 0xFFFFu)) * al);
        atomicAdd(&po[2 * idx2 + 1], bf2f((unsigned short)(a >> 16))     * al);
    }
}

// =====================================================================
// h = elu(agg + gat_bias) -> bf16
// =====================================================================
__global__ void elu_bias_kernel(const float* __restrict__ agg,
                                const float* __restrict__ gbias,
                                unsigned short* __restrict__ hbuf, int n)
{
    int t = blockIdx.x * blockDim.x + threadIdx.x;
    if (t >= n) return;
    float v = agg[t] + gbias[t & (HC - 1)];
    v = v > 0.0f ? v : __expf(v) - 1.0f;
    hbuf[t] = f2bf(v);
}

// =====================================================================
// GRU gates + relu -> d_out   (one thread per output element)
// =====================================================================
__global__ void gru_gate_kernel(const float* __restrict__ gi,
                                const float* __restrict__ gh,
                                const float* __restrict__ hprev,
                                float* __restrict__ outp, int n)
{
    int t = blockIdx.x * blockDim.x + threadIdx.x;
    if (t >= n) return;
    int m = t >> 8, j = t & 255;
    const float* gim = gi + (size_t)m * G3;
    const float* ghm = gh + (size_t)m * G3;
    float r  = sigmoidf(gim[j]       + ghm[j]);
    float z  = sigmoidf(gim[256 + j] + ghm[256 + j]);
    float nn = tanhf(gim[512 + j] + r * ghm[512 + j]);
    float nv = (1.0f - z) * nn + z * hprev[t];
    outp[t] = nv > 0.0f ? nv : 0.0f;
}

// =====================================================================
// host side
// =====================================================================
extern "C" void kernel_launch(void* const* d_in, const int* in_sizes, int n_in,
                              void* d_out, int out_size, void* d_ws, size_t ws_size,
                              hipStream_t stream) {
    const float* x       = (const float*)d_in[0];
    const float* outprev = (const float*)d_in[1];
    const int*   eidx    = (const int*)d_in[2];
    const float* Wl      = (const float*)d_in[3];
    const float* bl      = (const float*)d_in[4];
    const float* Wr      = (const float*)d_in[5];
    const float* br      = (const float*)d_in[6];
    const float* att     = (const float*)d_in[7];
    const float* gbias   = (const float*)d_in[8];
    const float* W_ih    = (const float*)d_in[9];
    const float* W_hh    = (const float*)d_in[10];
    const float* b_ih    = (const float*)d_in[11];
    const float* b_hh    = (const float*)d_in[12];
    const int* src = eidx;
    const int* dst = eidx + NEDGE;

    // ---- carve workspace (256B aligned)
    char* ws = (char*)d_ws;
    size_t off = 0;
    auto carve = [&](size_t bytes) -> char* {
        char* p = ws + off;
        off = (off + bytes + 255) & ~(size_t)255;
        return p;
    };
    unsigned short* xl_bf = (unsigned short*)carve((size_t)N_SRC * HC * 2);
    unsigned short* xr_bf = (unsigned short*)carve((size_t)N_DST * HC * 2);
    float* logits = (float*)carve((size_t)NEDGE * 4 * 4);
    float* mbuf   = (float*)carve((size_t)N_DST * 4 * 4);
    float* denom  = (float*)carve((size_t)N_DST * 4 * 4);
    float* agg    = (float*)carve((size_t)N_DST * HC * 4);
    unsigned short* hbuf = (unsigned short*)carve((size_t)N_DST * HC * 2);
    float* gi     = (float*)carve((size_t)N_DST * G3 * 4);
    float* gh     = (float*)carve((size_t)N_DST * G3 * 4);
    (void)ws_size; (void)in_sizes; (void)n_in; (void)out_size;

    // ---- init m / denom / agg
    {
        int nAgg = N_DST * HC, nMD = N_DST * 4;
        init_ws_kernel<<<(nAgg + 255) / 256, 256, 0, stream>>>(mbuf, denom, agg, nAgg, nMD);
    }
    // ---- xl = x @ Wl + bl   (bf16 out)
    gemm_tile<false, false, true><<<dim3((N_SRC + TM - 1) / TM, HC / TN), 256, 0, stream>>>(
        x, Wl, bl, xl_bf, N_SRC, HC, D_IN);
    // ---- xr = out @ Wr + br (bf16 out)
    gemm_tile<false, false, true><<<dim3(N_DST / TM, HC / TN), 256, 0, stream>>>(
        outprev, Wr, br, xr_bf, N_DST, HC, D_IN);
    // ---- attention logits + segment max
    edge_logits_kernel<<<NEDGE / 8, 256, 0, stream>>>(xl_bf, xr_bf, src, dst, att,
                                                      logits, mbuf, NEDGE);
    // ---- exp & segment denom
    edge_exp_kernel<<<(NEDGE * 4 + 255) / 256, 256, 0, stream>>>(dst, mbuf, logits,
                                                                 denom, NEDGE * 4);
    // ---- weighted aggregation
    edge_agg_kernel<<<NEDGE / 8, 256, 0, stream>>>(xl_bf, src, dst, logits, denom,
                                                   agg, NEDGE);
    // ---- h = elu(agg + gat_bias) -> bf16
    elu_bias_kernel<<<(N_DST * HC + 255) / 256, 256, 0, stream>>>(agg, gbias, hbuf,
                                                                  N_DST * HC);
    // ---- gi = h @ W_ih^T + b_ih  (f32 out)
    gemm_tile<true, true, false><<<dim3(N_DST / TM, G3 / TN), 256, 0, stream>>>(
        hbuf, W_ih, b_ih, gi, N_DST, G3, HC);
    // ---- gh = out @ W_hh^T + b_hh (f32 out)
    gemm_tile<false, true, false><<<dim3(N_DST / TM, G3 / TN), 256, 0, stream>>>(
        outprev, W_hh, b_hh, gh, N_DST, G3, D_IN);
    // ---- GRU gates + relu
    gru_gate_kernel<<<(N_DST * 256 + 255) / 256, 256, 0, stream>>>(
        gi, gh, outprev, (float*)d_out, N_DST * 256);
}
